// WindowAttention4d_82884278879126
// MI455X (gfx1250) — compile-verified, matrix operand order checked by reading, not run on hardware
//
#include <hip/hip_runtime.h>
#include <hip/hip_bf16.h>

typedef __attribute__((ext_vector_type(16))) _Float16 v16h;
typedef __attribute__((ext_vector_type(8)))  _Float16 v8h;
typedef __attribute__((ext_vector_type(8)))  float    v8f;
typedef __attribute__((ext_vector_type(4)))  float    v4f;

#define WSZ    7
#define N_TOK  49
#define NPAD   64
#define C_CH   128
#define HEADS  4
#define HD     32
#define OC3    384           // 3*C
#define CHW    (C_CH * N_TOK)

// Dynamic-LDS byte offsets (160 KB total -> 2 workgroups / 320KB WGP)
#define OFF_XT   0           // 64 x 128 f16 : x^T (reused later as O)
#define OFF_QK   16384       // 64 x 256 f16 : q (cols 0..127, pre-scaled), k (128..255)
#define OFF_VT   49152       // 128 x 64 f16 : v^T  (row = h*32+d, col = token m)
#define OFF_S    65536       // 4 x 64 x 64 f32 : attention scores
#define OFF_P    131072      // 4 x 64 x 64 f16 : softmax probs (pad cols zeroed)
#define LDS_BYTES 163840

// Load one 16x16x32 f16 WMMA operand fragment from a row-major matrix
// ("rows" = the 16 M-rows for A, or the 16 N-columns for B fetched from the
//  transposed storage). Matches the ISA 16-bit A/B layout:
//   lanes 0-15  : row r, K in {0..7} U {16..23}
//   lanes 16-31 : row r, K in {8..15} U {24..31}
__device__ __forceinline__ v16h load_frag(const _Float16* __restrict__ base,
                                          int pitch, int r0, int k0) {
  const int lane = threadIdx.x & 31;
  const _Float16* p = base + (r0 + (lane & 15)) * pitch + k0 + ((lane & 16) ? 8 : 0);
  v8h lo = *(const v8h*)p;
  v8h hi = *(const v8h*)(p + 16);
  v16h r;
#pragma unroll
  for (int i = 0; i < 8; ++i) { r[i] = lo[i]; r[i + 8] = hi[i]; }
  return r;
}

#define WMMA(A, B, C) \
  __builtin_amdgcn_wmma_f32_16x16x32_f16(false, (A), false, (B), (short)0, (C), false, false)

// ---------------------------------------------------------------------------
// One-time weight conversion f32 -> f16 into workspace (L2-resident afterwards)
// ---------------------------------------------------------------------------
__global__ void prep_weights(const float* __restrict__ qkv_w,
                             const float* __restrict__ proj_w,
                             _Float16* __restrict__ wbuf) {
  int i = blockIdx.x * blockDim.x + threadIdx.x;
  if (i < OC3 * C_CH) {
    wbuf[i] = (_Float16)qkv_w[i];
  } else if (i < OC3 * C_CH + C_CH * C_CH) {
    wbuf[i] = (_Float16)proj_w[i - OC3 * C_CH];
  }
}

// ---------------------------------------------------------------------------
// Fully fused window attention: one workgroup (8 wave32) per window.
// ---------------------------------------------------------------------------
__global__ __launch_bounds__(256)
void win_attn(const float* __restrict__ x, const float* __restrict__ mask,
              const _Float16* __restrict__ wqkv, const float* __restrict__ qkv_b,
              const _Float16* __restrict__ wproj, const float* __restrict__ proj_b,
              const float* __restrict__ bias_table, float* __restrict__ out) {
  extern __shared__ char smem[];
  _Float16* xt  = (_Float16*)(smem + OFF_XT);   // [n][c]  pitch 128
  _Float16* qk  = (_Float16*)(smem + OFF_QK);   // [n][o]  pitch 256 (o<128 q, else k)
  _Float16* vt  = (_Float16*)(smem + OFF_VT);   // [d][m]  pitch 64, d = h*32+dd
  float*    S   = (float*)   (smem + OFF_S);    // [h][n][m] pitch 64
  _Float16* P   = (_Float16*)(smem + OFF_P);    // [h][n][m] pitch 64
  _Float16* olds = xt;                          // reuse region: [n][c] pitch 128

  const int b    = blockIdx.x;
  const int t    = threadIdx.x;
  // Wave id is uniform across the wave: force it scalar so all tile-selection
  // control flow lowers to SALU + s_cbranch instead of v_cmpx/EXEC masking.
  const int wave = __builtin_amdgcn_readfirstlane(t >> 5);
  const int lane = t & 31;
  const int roff = (lane & 16) ? 8 : 0;         // D-matrix row offset for this half-wave
  const float scale = 0.17677669529663687f;     // 32^-0.5

  // ---- Phase 0: stage x^T into LDS as f16 (float4 loads), zero row pad ----
  for (int i = t; i < (NPAD - N_TOK) * C_CH; i += 256)
    xt[(N_TOK + i / C_CH) * C_CH + (i % C_CH)] = (_Float16)0.0f;
  const float* xb = x + (size_t)b * CHW;        // x[b][c][n], row-major over (c,n)
  for (int i4 = t; i4 < CHW / 4; i4 += 256) {
    v4f v = *(const v4f*)(xb + i4 * 4);
#pragma unroll
    for (int e = 0; e < 4; ++e) {
      int i = i4 * 4 + e;
      int c = i / N_TOK, n = i % N_TOK;
      xt[n * C_CH + c] = (_Float16)v[e];
    }
  }
  __syncthreads();

  // ---- Phase 1: QKV GEMM  (M=64, N=384, K=128) ----------------------------
  // Wave layout: m-row = wave&3 (fixed -> A fragments hoisted), o-tiles split
  // between the two waves sharing an m-row. q/k/v branch is uniform on o0.
  {
    const int m0 = (wave & 3) * 16;
    v16h ax[4];
#pragma unroll
    for (int kk = 0; kk < 4; ++kk) ax[kk] = load_frag(xt, C_CH, m0, kk * 32);

    for (int ot = (wave >> 2); ot < 24; ot += 2) {
      const int o0 = ot * 16;                   // scalar (wave-uniform)
      v8f acc = {};
#pragma unroll
      for (int kk = 0; kk < 4; ++kk) {
        v16h bf = load_frag(wqkv, C_CH, o0, kk * 32);  // W rows are the B columns
        acc = WMMA(ax[kk], bf, acc);
      }
      const int o = o0 + (lane & 15);
      const float bias = qkv_b[o];
      if (o0 < C_CH) {                          // q: pre-scale, row-major
#pragma unroll
        for (int j = 0; j < 8; ++j)
          qk[(m0 + roff + j) * 256 + o] = (_Float16)((acc[j] + bias) * scale);
      } else if (o0 < 2 * C_CH) {               // k: row-major
#pragma unroll
        for (int j = 0; j < 8; ++j)
          qk[(m0 + roff + j) * 256 + o] = (_Float16)(acc[j] + bias);
      } else {                                  // v: transposed, packed 16B store
        const int d = o - 2 * C_CH;
        v8h pk;
#pragma unroll
        for (int j = 0; j < 8; ++j) pk[j] = (_Float16)(acc[j] + bias);
        *(v8h*)(vt + d * NPAD + m0 + roff) = pk;
      }
    }
  }
  __syncthreads();

  // ---- Phase 2: S = q * k^T per head (K = hd = 32, one WMMA per tile) -----
  // Wave layout: head = wave>>1, two n-tiles per wave -> q fragments hoisted.
  {
    const int h = wave >> 1;
    const int ntb = (wave & 1) * 2;             // n-tile base: 0 or 2
    v16h aq[2];
#pragma unroll
    for (int i = 0; i < 2; ++i)
      aq[i] = load_frag(qk, 256, (ntb + i) * 16, h * HD);

    float* Sh = S + h * NPAD * NPAD;
#pragma unroll
    for (int mt = 0; mt < 4; ++mt) {
      const int m0 = mt * 16;
      v16h bk = load_frag(qk + C_CH + h * HD, 256, m0, 0);
      const int m = m0 + (lane & 15);
#pragma unroll
      for (int i = 0; i < 2; ++i) {
        v8f acc = {};
        acc = WMMA(aq[i], bk, acc);
        const int n0 = (ntb + i) * 16;
#pragma unroll
        for (int j = 0; j < 8; ++j) Sh[(n0 + roff + j) * NPAD + m] = acc[j];
      }
    }
  }
  __syncthreads();

  // ---- Phase 3: bias + mask + softmax (one thread per (head, token)) ------
  if (t < HEADS * N_TOK) {
    int h = t / N_TOK, n = t % N_TOK;
    float* row = S + h * NPAD * NPAD + n * NPAD;
    const float* mrow = mask + (size_t)(b & 63) * N_TOK * N_TOK + n * N_TOK;
    int i1 = n / WSZ, j1 = n % WSZ;
    float mx = -1e30f;
    for (int m = 0; m < N_TOK; ++m) {
      int i2 = m / WSZ, j2 = m % WSZ;
      int ridx = (i1 - i2 + WSZ - 1) * (2 * WSZ - 1) + (j1 - j2 + WSZ - 1);
      float v = row[m] + bias_table[ridx * HEADS + h] + mrow[m];
      row[m] = v;
      mx = fmaxf(mx, v);
    }
    float sum = 0.0f;
    for (int m = 0; m < N_TOK; ++m) { float e = __expf(row[m] - mx); row[m] = e; sum += e; }
    float rs = 1.0f / sum;
    _Float16* prow = P + h * NPAD * NPAD + n * NPAD;
    for (int m = 0; m < N_TOK; ++m) prow[m] = (_Float16)(row[m] * rs);
    for (int m = N_TOK; m < NPAD; ++m) prow[m] = (_Float16)0.0f;  // kill pad cols
  }
  __syncthreads();

  // ---- Phase 4: O = P * v per head (K = 64, two WMMA steps) ---------------
  // Wave layout: head = wave>>1, d-tile = wave&1 -> v^T fragments hoisted.
  {
    const int h = wave >> 1;
    const int d0 = (wave & 1) * 16;
    v16h bv[2];
#pragma unroll
    for (int kk = 0; kk < 2; ++kk)
      bv[kk] = load_frag(vt + h * HD * NPAD, NPAD, d0, kk * 32);

    const _Float16* Ph = P + h * NPAD * NPAD;
    const int d = d0 + (lane & 15);
#pragma unroll
    for (int nt = 0; nt < 4; ++nt) {
      const int n0 = nt * 16;
      v8f acc = {};
#pragma unroll
      for (int kk = 0; kk < 2; ++kk) {
        v16h a = load_frag(Ph, NPAD, n0, kk * 32);
        acc = WMMA(a, bv[kk], acc);
      }
#pragma unroll
      for (int j = 0; j < 8; ++j)
        olds[(n0 + roff + j) * C_CH + h * HD + d] = (_Float16)acc[j];
    }
  }
  __syncthreads();

  // ---- Phase 5: output projection (M=64, N=128, K=128) + store ------------
  // Wave layout: c-tile = wave -> proj-weight fragments hoisted (global, L2-hot).
  {
    const int c0 = wave * 16;
    v16h bw[4];
#pragma unroll
    for (int kk = 0; kk < 4; ++kk) bw[kk] = load_frag(wproj, C_CH, c0, kk * 32);

    const int co = c0 + (lane & 15);
    const float pb = proj_b[co];
    float* orow = out + (size_t)b * CHW + co * N_TOK;   // out[b][co][n]
#pragma unroll
    for (int nt = 0; nt < 4; ++nt) {
      const int n0 = nt * 16;
      v8f acc = {};
#pragma unroll
      for (int kk = 0; kk < 4; ++kk) {
        v16h a = load_frag(olds, C_CH, n0, kk * 32);
        acc = WMMA(a, bw[kk], acc);
      }
#pragma unroll
      for (int j = 0; j < 8; ++j) {
        int n = n0 + roff + j;
        if (n < N_TOK) orow[n] = acc[j] + pb;   // fixed co per lane, contiguous n
      }
    }
  }
}

extern "C" void kernel_launch(void* const* d_in, const int* in_sizes, int n_in,
                              void* d_out, int out_size, void* d_ws, size_t ws_size,
                              hipStream_t stream) {
  const float* x          = (const float*)d_in[0];
  const float* mask       = (const float*)d_in[1];
  const float* qkv_w      = (const float*)d_in[2];
  const float* qkv_b      = (const float*)d_in[3];
  const float* proj_w     = (const float*)d_in[4];
  const float* proj_b     = (const float*)d_in[5];
  const float* bias_table = (const float*)d_in[6];

  _Float16* wbuf = (_Float16*)d_ws;                 // [qkv_w f16 | proj_w f16]
  const int wtotal = OC3 * C_CH + C_CH * C_CH;      // 65536 halves = 128 KB
  prep_weights<<<(wtotal + 255) / 256, 256, 0, stream>>>(qkv_w, proj_w, wbuf);

  // 160 KB dynamic LDS > default cap: raise the limit (idempotent, capture-safe).
  (void)hipFuncSetAttribute((const void*)win_attn,
                            hipFuncAttributeMaxDynamicSharedMemorySize, LDS_BYTES);

  win_attn<<<4096, 256, LDS_BYTES, stream>>>(x, mask, wbuf, qkv_b,
                                             wbuf + OC3 * C_CH, proj_b,
                                             bias_table, (float*)d_out);
}
